// FFNPCCNorm_29291676959209
// MI455X (gfx1250) — matrix operand order repaired and checked by reading, not compiled, not run on hardware
//
#include <hip/hip_runtime.h>
#include <hip/hip_bf16.h>

// ---------------------------------------------------------------------------
// FFN PCC-norm: out[n,o] = max( (xc[n,:]·wc[o,:]) / (|xc[n]|·|wc[o]|), 1e-10 ) + bias[o]
// N=262144, IN=OUT=256.
// HBM roofline: 0.54 GB @ 23.3 TB/s ~= 22 us. Split-bf16 (hi/lo) WMMA GEMM gives
// ~1e-5 accuracy using the compute headroom. Normalizations are folded into the
// stored operands (xn = xc/|xc|, wn = wc/|wc|) so WMMA output == correlation.
// Each B fragment is loaded once per wave and reused across 4 M-tiles (~1 GB of
// L2-resident B traffic). Per kb-step all 16 A-fragment ds_load_b128 are issued
// as one clause from a single opaque byte offset (immediate ds offsets), then
// 12 WMMAs run back-to-back: no VALU writes between WMMAs -> no hazard v_nops,
// and LICM can't hoist+spill the fragments. __launch_bounds__(128,2) keeps
// >=2 waves/SIMD (2 WGs/WGP at 64 KB LDS each).
// ---------------------------------------------------------------------------

typedef __attribute__((ext_vector_type(16))) __bf16 v16bf;
typedef __attribute__((ext_vector_type(8)))  __bf16 v8bf;
typedef __attribute__((ext_vector_type(8)))  float  v8f;
typedef __attribute__((ext_vector_type(4)))  float  v4f;

#define NROWS     262144
#define KDIM      256
#define ODIM      256
#define TILE_ROWS 64      // rows per workgroup (4 M-tiles shared by all waves)
#define THREADS   128     // 4 wave32 waves; each wave owns 4 N-tiles
#define XLO_ELTS  (TILE_ROWS * KDIM)   // lo-plane element offset inside xs

__device__ __forceinline__ float wave_reduce_sum(float v) {
#pragma unroll
    for (int off = 16; off > 0; off >>= 1)
        v += __shfl_xor(v, off, 32);
    return v;
}

// One wave per weight row: center, normalize by ||wc||, emit split-bf16 row
// (row-major == B-fragment per-lane layout: lane n reads two runs of 8 K halfs per K32 block).
__global__ __launch_bounds__(32)
void prep_weight(const float* __restrict__ w,
                 __bf16* __restrict__ whi,
                 __bf16* __restrict__ wlo) {
    const int o    = blockIdx.x;
    const int lane = threadIdx.x;
    const float* row = w + (size_t)o * KDIM + lane * 8;
    v4f f0 = *(const v4f*)(row);
    v4f f1 = *(const v4f*)(row + 4);

    float s = 0.f;
#pragma unroll
    for (int j = 0; j < 4; ++j) { s += f0[j]; s += f1[j]; }
    s = wave_reduce_sum(s);
    const float mean = s * (1.0f / (float)KDIM);

    float q = 0.f;
#pragma unroll
    for (int j = 0; j < 4; ++j) {
        float a = f0[j] - mean, b = f1[j] - mean;
        q += a * a + b * b;
    }
    q = wave_reduce_sum(q);
    const float rn = rsqrtf(q);   // 1/w_len

    v8bf hv, lv;
#pragma unroll
    for (int j = 0; j < 4; ++j) {
        float a = (f0[j] - mean) * rn, b = (f1[j] - mean) * rn;
        __bf16 ah = (__bf16)a, bh = (__bf16)b;
        hv[j] = ah;           hv[j + 4] = bh;
        lv[j] = (__bf16)(a - (float)ah);
        lv[j + 4] = (__bf16)(b - (float)bh);
    }
    *(v8bf*)(whi + (size_t)o * KDIM + lane * 8) = hv;
    *(v8bf*)(wlo + (size_t)o * KDIM + lane * 8) = lv;
}

__global__ __launch_bounds__(THREADS, 2)   // >=2 waves/SIMD, spill-free
void pcc_gemm(const float* __restrict__ x,
              const __bf16* __restrict__ whi,
              const __bf16* __restrict__ wlo,
              const float* __restrict__ bias,
              float* __restrict__ out) {
    // hi plane: elements [0, 16384); lo plane: [16384, 32768). 64 KB total.
    __shared__ __bf16 xs[2 * TILE_ROWS * KDIM];

    const int tid  = threadIdx.x;
    const int wave = tid >> 5;
    const int lane = tid & 31;
    const int row0 = blockIdx.x * TILE_ROWS;

    // Warm L2 with the split weights (global_prefetch_b8).
    for (int i = tid; i < (ODIM * KDIM) / 64; i += THREADS) {
        __builtin_prefetch(whi + i * 64, 0, 0);
        __builtin_prefetch(wlo + i * 64, 0, 0);
    }

    // -------- Phase 1: per-row center+normalize, split-bf16 tile into LDS ----
    // 4 waves x 16 rows = 64 rows; loads batched 4 rows deep to overlap latency.
#pragma unroll 1
    for (int ii = 0; ii < 4; ++ii) {
        const int rbase = wave * 16 + ii * 4;
        v4f f0[4], f1[4];
#pragma unroll
        for (int u = 0; u < 4; ++u) {
            const float* rp = x + (size_t)(row0 + rbase + u) * KDIM + lane * 8;
            f0[u] = *(const v4f*)(rp);
            f1[u] = *(const v4f*)(rp + 4);
        }
#pragma unroll
        for (int u = 0; u < 4; ++u) {
            const int rl = rbase + u;
            float s = 0.f;
#pragma unroll
            for (int j = 0; j < 4; ++j) { s += f0[u][j]; s += f1[u][j]; }
            s = wave_reduce_sum(s);
            const float mean = s * (1.0f / (float)KDIM);

            float q = 0.f;
#pragma unroll
            for (int j = 0; j < 4; ++j) {
                float a = f0[u][j] - mean, b = f1[u][j] - mean;
                q += a * a + b * b;
            }
            q = wave_reduce_sum(q);
            const float rn = rsqrtf(q);   // 1/x_len

            v8bf hv, lv;
#pragma unroll
            for (int j = 0; j < 4; ++j) {
                float a = (f0[u][j] - mean) * rn, b = (f1[u][j] - mean) * rn;
                __bf16 ah = (__bf16)a, bh = (__bf16)b;
                hv[j] = ah;           hv[j + 4] = bh;
                lv[j] = (__bf16)(a - (float)ah);
                lv[j + 4] = (__bf16)(b - (float)bh);
            }
            *(v8bf*)(&xs[rl * KDIM + lane * 8]) = hv;
            *(v8bf*)(&xs[XLO_ELTS + rl * KDIM + lane * 8]) = lv;
        }
    }
    __syncthreads();

    // -------- Phase 2: split-bf16 WMMA GEMM with 4x B-fragment reuse --------
    // Wave w owns N-tiles [4w, 4w+4); every wave touches all 4 M-tiles, so each
    // B fragment (loaded from L2-resident wn) feeds 4 M-tiles x 3 WMMAs.
    const int h  = lane >> 4;    // K-half selector
    const int nl = lane & 15;    // row (A) / column (B,D) within tile

#pragma unroll 1
    for (int t = 0; t < 4; ++t) {
        const int nt = wave * 4 + t;
        const int o  = nt * 16 + nl;
        const float bo = bias[o];

        v8f acc[4] = {};
        // B 32x16 layout: lane = column o, K halfs [8h,8h+8) and [16+8h,16+8h+8) of block 32kb.
        const __bf16* bph = whi + (size_t)o * KDIM + 8 * h;
        const __bf16* bpl = wlo + (size_t)o * KDIM + 8 * h;
#pragma unroll
        for (int kb = 0; kb < 8; ++kb) {
            v8bf b0 = *(const v8bf*)(bph + kb * 32);
            v8bf b1 = *(const v8bf*)(bph + kb * 32 + 16);
            v8bf c0 = *(const v8bf*)(bpl + kb * 32);
            v8bf c1 = *(const v8bf*)(bpl + kb * 32 + 16);
            v16bf bh, bl;
#pragma unroll
            for (int j = 0; j < 8; ++j) {
                bh[j] = b0[j]; bh[j + 8] = b1[j];
                bl[j] = c0[j]; bl[j + 8] = c1[j];
            }

            // One opaque BYTE offset per kb; all 16 A-fragment ds_load_b128
            // derive from it via immediate ds offsets (mt stride 8192 B,
            // lo plane +32768 B, second K-chunk +32 B). Batched as one clause,
            // no VALU between the WMMAs below -> no WMMA hazard v_nops.
            int ab = (nl * KDIM + kb * 32 + 8 * h) * (int)sizeof(__bf16);
            asm volatile("" : "+v"(ab));
            const char* bb = (const char*)xs + ab;

            v16bf ahi[4], alo[4];
#pragma unroll
            for (int mt = 0; mt < 4; ++mt) {
                // 16-bit A 16x32 layout: lane half h holds K in [8h,8h+8) and
                // [16+8h,16+8h+8) of this K32 block.
                v8bf a0 = *(const v8bf*)(bb + mt * 8192);
                v8bf a1 = *(const v8bf*)(bb + mt * 8192 + 32);
                v8bf d0 = *(const v8bf*)(bb + mt * 8192 + 32768);
                v8bf d1 = *(const v8bf*)(bb + mt * 8192 + 32800);
#pragma unroll
                for (int j = 0; j < 8; ++j) {
                    ahi[mt][j] = a0[j]; ahi[mt][j + 8] = a1[j];
                    alo[mt][j] = d0[j]; alo[mt][j + 8] = d1[j];
                }
            }
#pragma unroll
            for (int mt = 0; mt < 4; ++mt) {
                acc[mt] = __builtin_amdgcn_wmma_f32_16x16x32_bf16(
                    false, ahi[mt], false, bh, (short)0, acc[mt], false, false);
                acc[mt] = __builtin_amdgcn_wmma_f32_16x16x32_bf16(
                    false, ahi[mt], false, bl, (short)0, acc[mt], false, false);
                acc[mt] = __builtin_amdgcn_wmma_f32_16x16x32_bf16(
                    false, alo[mt], false, bh, (short)0, acc[mt], false, false);
            }
        }

        // acc == correlation (normalizations pre-folded); clamp + bias, store.
#pragma unroll
        for (int mt = 0; mt < 4; ++mt) {
#pragma unroll
            for (int j = 0; j < 8; ++j) {
                const int ml = mt * 16 + 8 * h + j;
                out[(size_t)(row0 + ml) * ODIM + o] = fmaxf(acc[mt][j], 1e-10f) + bo;
            }
        }
    }
}

extern "C" void kernel_launch(void* const* d_in, const int* in_sizes, int n_in,
                              void* d_out, int out_size, void* d_ws, size_t ws_size,
                              hipStream_t stream) {
    const float* x    = (const float*)d_in[0];   // [N, 256]
    const float* w    = (const float*)d_in[1];   // [256, 256]
    const float* bias = (const float*)d_in[2];   // [256]
    float* out = (float*)d_out;

    const size_t wbytes = (size_t)ODIM * KDIM * sizeof(__bf16);   // 128 KB
    __bf16* whi = (__bf16*)d_ws;
    __bf16* wlo = (__bf16*)((char*)d_ws + wbytes);

    prep_weight<<<ODIM, 32, 0, stream>>>(w, whi, wlo);
    pcc_gemm<<<NROWS / TILE_ROWS, THREADS, 0, stream>>>(x, whi, wlo, bias, out);
}
